// Generator_12756052869773
// MI455X (gfx1250) — matrix operand order repaired and checked by reading, not compiled, not run on hardware
//
#include <hip/hip_runtime.h>
#include <hip/hip_bf16.h>

typedef __attribute__((ext_vector_type(16))) __bf16 v16bf;
typedef __attribute__((ext_vector_type(8)))  __bf16 v8bf;
typedef __attribute__((ext_vector_type(8)))  float  v8f;

#define NEG_ATT  0.2f
#define NEG_HEAD 0.01f
#define WT_STRIDE 136   // padded row stride (elements) for transposed W in LDS

__device__ __forceinline__ __bf16 f2bf(float f) { return (__bf16)f; }

// guaranteed-native fp32 atomic add (global_atomic_add_f32, no CAS loop)
__device__ __forceinline__ void atomAddF(float* p, float v) {
  unsafeAtomicAdd(p, v);
}

// order-preserving float <-> uint mapping for atomicMax on floats
__device__ __forceinline__ unsigned fmap(float f) {
  unsigned u = __float_as_uint(f);
  return (u & 0x80000000u) ? ~u : (u | 0x80000000u);
}
__device__ __forceinline__ float funmap(unsigned u) {
  unsigned v = (u & 0x80000000u) ? (u & 0x7FFFFFFFu) : ~u;
  return __uint_as_float(v);
}

// ---------------- GEMM: H[nrows,128] = X[nrows,128] @ W[128,128] ----------------
// bf16 WMMA, f32 accumulate. 256 threads = 8 waves, each wave owns a 16x128 tile.
// W is transposed into LDS (Wt[n][k]) so every B fragment is two runs of 16 contiguous bytes.
__global__ __launch_bounds__(256) void k_gemm(const float* __restrict__ X,
                                              const float* __restrict__ W,
                                              float* __restrict__ H, int nrows) {
  __shared__ __align__(16) __bf16 Wt[128 * WT_STRIDE];
  const int tid = threadIdx.x;
  // cooperative load + transpose: Wt[n*WT_STRIDE + k] = W[k*128 + n]
  for (int i = tid; i < 128 * 128; i += 256) {
    const int k = i >> 7, n = i & 127;          // consecutive tid -> consecutive n (coalesced)
    Wt[n * WT_STRIDE + k] = f2bf(W[i]);
  }
  __syncthreads();

  const int lane = tid & 31;
  const int wave = tid >> 5;
  const int hi   = lane >> 4;      // lane half
  const int lo   = lane & 15;
  const int rowBase = blockIdx.x * 128 + wave * 16;

  v8f acc[8];
#pragma unroll
  for (int t = 0; t < 8; ++t)
#pragma unroll
    for (int r = 0; r < 8; ++r) acc[t][r] = 0.0f;

  const int  arow  = rowBase + lo;
  const bool av    = arow < nrows;
  const int  arowc = av ? arow : (nrows - 1);
  const float scale = av ? 1.0f : 0.0f;
  const float* xr = X + (size_t)arowc * 128;

#pragma unroll
  for (int ks = 0; ks < 4; ++ks) {
    const int k0 = ks * 32;
    // ---- A fragment: elems 0-7 -> K = k0+8*hi+0..7, elems 8-15 -> K = k0+16+8*hi+0..7
    const float4* c0p = (const float4*)(xr + k0 + 8 * hi);
    const float4* c1p = (const float4*)(xr + k0 + 16 + 8 * hi);
    float4 c0 = c0p[0], c1 = c0p[1], c2 = c1p[0], c3 = c1p[1];
    v16bf a;
    a[0]  = f2bf(c0.x * scale); a[1]  = f2bf(c0.y * scale);
    a[2]  = f2bf(c0.z * scale); a[3]  = f2bf(c0.w * scale);
    a[4]  = f2bf(c1.x * scale); a[5]  = f2bf(c1.y * scale);
    a[6]  = f2bf(c1.z * scale); a[7]  = f2bf(c1.w * scale);
    a[8]  = f2bf(c2.x * scale); a[9]  = f2bf(c2.y * scale);
    a[10] = f2bf(c2.z * scale); a[11] = f2bf(c2.w * scale);
    a[12] = f2bf(c3.x * scale); a[13] = f2bf(c3.y * scale);
    a[14] = f2bf(c3.z * scale); a[15] = f2bf(c3.w * scale);

    // ---- 8 column tiles of B: elems 0-7 -> K = k0+8*hi+0..7, elems 8-15 -> K = k0+16+8*hi+0..7, N = t*16+lo ----
#pragma unroll
    for (int t = 0; t < 8; ++t) {
      const int n = t * 16 + lo;
      const __bf16* wp = &Wt[(size_t)n * WT_STRIDE + k0 + 8 * hi];
      v8bf blo = *(const v8bf*)(wp);
      v8bf bhi = *(const v8bf*)(wp + 16);
      v16bf b = __builtin_shufflevector(blo, bhi, 0, 1, 2, 3, 4, 5, 6, 7,
                                        8, 9, 10, 11, 12, 13, 14, 15);
      acc[t] = __builtin_amdgcn_wmma_f32_16x16x32_bf16(false, a, false, b,
                                                       (short)0, acc[t], false, false);
    }
  }
  // ---- store C: VGPR r -> M = r + 8*hi, N = lane&15 ----
#pragma unroll
  for (int t = 0; t < 8; ++t) {
    const int col = t * 16 + lo;
#pragma unroll
    for (int r = 0; r < 8; ++r) {
      const int row = rowBase + r + 8 * hi;
      if (row < nrows) H[(size_t)row * 128 + col] = acc[t][r];
    }
  }
}

// ---------------- alpha_s / alpha_d: per-node dot with attention vectors --------
__global__ __launch_bounds__(256) void k_alpha(const float* __restrict__ H,
                                               const float* __restrict__ avs,
                                               const float* __restrict__ avd,
                                               float* __restrict__ as,
                                               float* __restrict__ ad, int n) {
  const int gw   = (blockIdx.x * 256 + threadIdx.x) >> 5;
  const int lane = threadIdx.x & 31;
  if (gw >= n) return;
  const float* h = H + (size_t)gw * 128;
  float ss = 0.f, sd = 0.f;
#pragma unroll
  for (int j = 0; j < 4; ++j) {
    const int c = lane + 32 * j;
    const float v = h[c];
    ss += v * avs[c];
    sd += v * avd[c];
  }
  for (int o = 16; o; o >>= 1) {
    ss += __shfl_down(ss, o, 32);
    sd += __shfl_down(sd, o, 32);
  }
  if (lane == 0) { as[gw] = ss; ad[gw] = sd; }
}

// fused per-layer init: m = -inf (ordered-uint 0), denom = 0, out = bias broadcast
__global__ void k_init(unsigned* __restrict__ m, float* __restrict__ den,
                       float* __restrict__ out, const float* __restrict__ b, int n) {
  const int i = blockIdx.x * 256 + threadIdx.x;
  const int total = n * 128;
  if (i < n) { m[i] = 0u; den[i] = 0.f; }
  for (int j = i; j < total; j += gridDim.x * 256) out[j] = b[j & 127];
}

// e = leaky(as[src]+ad[dst]); segment max via ordered-uint atomicMax
__global__ void k_edge_max(const int* __restrict__ ei, int E, int n,
                           const float* __restrict__ as, const float* __restrict__ ad,
                           float* __restrict__ eb, unsigned* __restrict__ m) {
  const int t = blockIdx.x * 256 + threadIdx.x;
  const int EL = E + n;
  if (t >= EL) return;
  int s, d;
  if (t < E) { s = ei[t]; d = ei[E + t]; } else { s = d = t - E; }
  float e = as[s] + ad[d];
  e = e >= 0.f ? e : NEG_ATT * e;
  eb[t] = e;
  atomicMax(&m[d], fmap(e));
}

// ex = exp(e - m[dst]); denom[dst] += ex   (ex overwrites e in-place)
__global__ void k_edge_exp(const int* __restrict__ ei, int E, int n,
                           float* __restrict__ eb, const unsigned* __restrict__ m,
                           float* __restrict__ den) {
  const int t = blockIdx.x * 256 + threadIdx.x;
  const int EL = E + n;
  if (t >= EL) return;
  const int d = (t < E) ? ei[E + t] : (t - E);
  const float ex = __expf(eb[t] - funmap(m[d]));
  eb[t] = ex;
  atomAddF(&den[d], ex);
}

// out[dst,:] += (ex/denom[dst]) * h[src,:]  — one wave per edge, float4 per lane
__global__ __launch_bounds__(256) void k_agg(const int* __restrict__ ei, int E, int n,
                                             const float* __restrict__ eb,
                                             const float* __restrict__ den,
                                             const float* __restrict__ H,
                                             float* __restrict__ out) {
  const long long gt = (long long)blockIdx.x * 256 + threadIdx.x;
  const int gw   = (int)(gt >> 5);
  const int lane = threadIdx.x & 31;
  const int EL = E + n;
  if (gw >= EL) return;
  int s, d;
  if (gw < E) { s = ei[gw]; d = ei[E + gw]; } else { s = d = gw - E; }
  const float w = eb[gw] / (den[d] + 1e-16f);
  const float4 v = ((const float4*)(H + (size_t)s * 128))[lane];
  float* o = out + (size_t)d * 128 + lane * 4;
  atomAddF(o + 0, w * v.x);
  atomAddF(o + 1, w * v.y);
  atomAddF(o + 2, w * v.z);
  atomAddF(o + 3, w * v.w);
}

__global__ void k_relu(float* __restrict__ z, int total) {
  const int i = blockIdx.x * 256 + threadIdx.x;
  if (i < total) z[i] = fmaxf(z[i], 0.f);
}

// y[i] = leaky(Z[idx[i],:] . Wv + b, 0.01) — one wave per row
__global__ __launch_bounds__(256) void k_head(const float* __restrict__ Z,
                                              const int* __restrict__ idx,
                                              const float* __restrict__ Wv,
                                              const float* __restrict__ bv,
                                              float* __restrict__ y, int cnt) {
  const int gw   = (blockIdx.x * 256 + threadIdx.x) >> 5;
  const int lane = threadIdx.x & 31;
  if (gw >= cnt) return;
  const float* z = Z + (size_t)idx[gw] * 128;
  float s = 0.f;
#pragma unroll
  for (int j = 0; j < 4; ++j) {
    const int c = lane + 32 * j;
    s += z[c] * Wv[c];
  }
  for (int o = 16; o; o >>= 1) s += __shfl_down(s, o, 32);
  if (lane == 0) {
    float v = s + bv[0];
    y[gw] = v >= 0.f ? v : NEG_HEAD * v;
  }
}

extern "C" void kernel_launch(void* const* d_in, const int* in_sizes, int n_in,
                              void* d_out, int out_size, void* d_ws, size_t ws_size,
                              hipStream_t stream) {
  const float* x      = (const float*)d_in[0];
  const int*   ei     = (const int*)d_in[1];
  const float* fx     = (const float*)d_in[2];
  const int*   fei    = (const int*)d_in[3];
  const int*   treat  = (const int*)d_in[4];
  const int*   ctrl   = (const int*)d_in[5];
  const float* W1     = (const float*)d_in[6];
  const float* as1    = (const float*)d_in[7];
  const float* ad1    = (const float*)d_in[8];
  const float* b1     = (const float*)d_in[9];
  const float* W2     = (const float*)d_in[10];
  const float* as2    = (const float*)d_in[11];
  const float* ad2    = (const float*)d_in[12];
  const float* b2     = (const float*)d_in[13];
  const float* Wy1    = (const float*)d_in[14];
  const float* by1    = (const float*)d_in[15];
  const float* Wy0    = (const float*)d_in[16];
  const float* by0    = (const float*)d_in[17];

  const int n  = in_sizes[0] / 128;
  const int E  = in_sizes[1] / 2;
  const int nt = in_sizes[4];
  const int nc = in_sizes[5];
  const int EL = E + n;

  // ---- carve workspace ----
  char* p = (char*)d_ws;
  auto alloc = [&](size_t bytes) -> void* {
    void* r = (void*)p;
    p += (bytes + 255) & ~(size_t)255;
    return r;
  };
  float*    h   = (float*)alloc((size_t)n * 128 * 4);
  float*    z1  = (float*)alloc((size_t)n * 128 * 4);
  float*    as  = (float*)alloc((size_t)n * 4);
  float*    ad  = (float*)alloc((size_t)n * 4);
  unsigned* mbf = (unsigned*)alloc((size_t)n * 4);
  float*    den = (float*)alloc((size_t)n * 4);
  float*    eb  = (float*)alloc((size_t)EL * 4);

  float* out = (float*)d_out;
  float* y1  = out;
  float* yc0 = out + nt;
  float* y0  = out + nt + nt;
  float* yc1 = out + nt + nt + nc;
  float* xZ2  = out + nt + nt + nc + nc;
  float* xfZ2 = xZ2 + (size_t)n * 128;

  const int gGemm  = (n + 127) / 128;
  const int gFeat  = (n * 128 + 255) / 256;
  const int gNodeW = (n * 32 + 255) / 256;          // wave-per-node
  const int gEdge  = (EL + 255) / 256;
  const int gEdgeW = (int)(((long long)EL * 32 + 255) / 256);  // wave-per-edge

  // one GATConv layer: input Xin -> output Zout (= segsum + bias)
  auto gat = [&](const float* Xin, const int* e, const float* Wp,
                 const float* avs, const float* avd, const float* bp, float* Zout) {
    k_gemm<<<gGemm, 256, 0, stream>>>(Xin, Wp, h, n);
    k_alpha<<<gNodeW, 256, 0, stream>>>(h, avs, avd, as, ad, n);
    k_init<<<gFeat, 256, 0, stream>>>(mbf, den, Zout, bp, n);
    k_edge_max<<<gEdge, 256, 0, stream>>>(e, E, n, as, ad, eb, mbf);
    k_edge_exp<<<gEdge, 256, 0, stream>>>(e, E, n, eb, mbf, den);
    k_agg<<<gEdgeW, 256, 0, stream>>>(e, E, n, eb, den, h, Zout);
  };

  // real graph
  gat(x, ei, W1, as1, ad1, b1, z1);
  k_relu<<<gFeat, 256, 0, stream>>>(z1, n * 128);
  gat(z1, ei, W2, as2, ad2, b2, xZ2);

  // fake graph
  gat(fx, fei, W1, as1, ad1, b1, z1);
  k_relu<<<gFeat, 256, 0, stream>>>(z1, n * 128);
  gat(z1, fei, W2, as2, ad2, b2, xfZ2);

  // heads
  const int gHt = (nt * 32 + 255) / 256;
  const int gHc = (nc * 32 + 255) / 256;
  k_head<<<gHt, 256, 0, stream>>>(xZ2,  treat, Wy1, by1, y1,  nt);
  k_head<<<gHt, 256, 0, stream>>>(xfZ2, treat, Wy0, by0, yc0, nt);
  k_head<<<gHc, 256, 0, stream>>>(xZ2,  ctrl,  Wy0, by0, y0,  nc);
  k_head<<<gHc, 256, 0, stream>>>(xfZ2, ctrl,  Wy1, by1, yc1, nc);
}
